// UserNeighborAggregator_42545946034759
// MI455X (gfx1250) — matrix operand order repaired and checked by reading, not compiled, not run on hardware
//
#include <hip/hip_runtime.h>
#include <hip/hip_bf16.h>
#include <math.h>

// ---------------- problem constants ----------------
#define B_USERS 1024
#define N_CACHE 100000
#define D_DIM   512
#define K_NEIGH 20
#define NCHUNKS 25
#define CHUNK   4000                  // N_CACHE / NCHUNKS
#define MROWS   64                    // user rows per sim-block
#define MGROUPS (B_USERS / MROWS)     // 16
#define CAPC    128                   // candidate buffer per row
#define ASTRIDE 520                   // LDS bf16 row stride (elems) for fallback kernel
#define AROWB   1040                  // LDS bytes per A row (65 x 16B chunks) in bf16 kernel
#define NEG_INF (-3.0e38f)

typedef __attribute__((ext_vector_type(16))) __bf16         v16bf;
typedef __attribute__((ext_vector_type(16))) unsigned short v16us;
typedef __attribute__((ext_vector_type(8)))  unsigned short v8us;
typedef __attribute__((ext_vector_type(8)))  float          v8f;
typedef __attribute__((ext_vector_type(2)))  float          v2f;

static __device__ __forceinline__ unsigned short f2bf(float f) {
  unsigned u = __builtin_bit_cast(unsigned, f);
  u += 0x7fffu + ((u >> 16) & 1u);              // round-to-nearest-even
  return (unsigned short)(u >> 16);
}

// ---------------- kernel 1: inverse L2 norms ----------------
__global__ void __launch_bounds__(256) invnorm_kernel(
    const float* __restrict__ src, float* __restrict__ inv, int rows) {
  const int lane = threadIdx.x & 31;
  const int row  = blockIdx.x * 8 + (threadIdx.x >> 5);
  if (row >= rows) return;
  const float* p = src + (size_t)row * D_DIM;
  float s = 0.f;
#pragma unroll
  for (int t = 0; t < 4; ++t) {
    float4 v = *(const float4*)(p + t * 128 + lane * 4);
    s += v.x * v.x + v.y * v.y + v.z * v.z + v.w * v.w;
  }
#pragma unroll
  for (int off = 16; off > 0; off >>= 1) s += __shfl_xor(s, off, 32);
  if (lane == 0) inv[row] = 1.f / (sqrtf(s) + 1e-12f);
}

// ---------------- kernel 1b: f32 -> bf16 bulk convert ----------------
__global__ void __launch_bounds__(256) cvt_bf16_kernel(
    const float* __restrict__ s, unsigned short* __restrict__ d, long n) {
  long i = ((long)blockIdx.x * 256 + threadIdx.x) * 4;
  if (i >= n) return;
  float4 f = *(const float4*)(s + i);
  uint2 p;
  p.x = (unsigned)f2bf(f.x) | ((unsigned)f2bf(f.y) << 16);
  p.y = (unsigned)f2bf(f.z) | ((unsigned)f2bf(f.w) << 16);
  *(uint2*)(d + i) = p;
}

// ---------------- top-k compaction (wave-parallel argmax rounds) ----------------
static __device__ __forceinline__ void compact_rows(
    float* bufv, int* bufi, float* thr, int* cnt, int wave, int lane) {
  for (int r = wave; r < MROWS; r += 8) {
    const int cc = min(cnt[r], CAPC);
    float lv[CAPC / 32]; int li[CAPC / 32];
#pragma unroll
    for (int t = 0; t < CAPC / 32; ++t) {
      int p = lane + 32 * t;
      bool ok = p < cc;
      lv[t] = ok ? bufv[r * CAPC + p] : NEG_INF;
      li[t] = ok ? bufi[r * CAPC + p] : -1;
    }
    float keepv = NEG_INF; int keepi = -1; float lastv = NEG_INF;
    for (int k = 0; k < K_NEIGH; ++k) {
      float bv = NEG_INF; int bt = -1;
#pragma unroll
      for (int t = 0; t < CAPC / 32; ++t) { if (lv[t] > bv) { bv = lv[t]; bt = t; } }
      float rv = bv; int rl = lane;
#pragma unroll
      for (int off = 16; off > 0; off >>= 1) {
        float ov = __shfl_xor(rv, off, 32);
        int   ol = __shfl_xor(rl, off, 32);
        if (ov > rv || (ov == rv && ol < rl)) { rv = ov; rl = ol; }
      }
      int mi = -1;
#pragma unroll
      for (int t = 0; t < CAPC / 32; ++t) { if (t == bt) mi = li[t]; }
      int gi = __shfl(mi, rl, 32);
      if (lane == rl) {
#pragma unroll
        for (int t = 0; t < CAPC / 32; ++t) { if (t == bt) lv[t] = NEG_INF; }
      }
      if (lane == k) { keepv = rv; keepi = gi; }
      lastv = rv;
    }
    if (lane < K_NEIGH) { bufv[r * CAPC + lane] = keepv; bufi[r * CAPC + lane] = keepi; }
    if (lane == 0) { cnt[r] = K_NEIGH; thr[r] = lastv; }
  }
}

// ---------------- shared LDS layout for both sims kernels ----------------
#define SMEM_A_BYTES  (MROWS * AROWB)                       // 66560 (== MROWS*ASTRIDE*2)
#define SMEM_BUFV_OFF (SMEM_A_BYTES)
#define SMEM_BUFI_OFF (SMEM_BUFV_OFF + MROWS * CAPC * 4)
#define SMEM_THR_OFF  (SMEM_BUFI_OFF + MROWS * CAPC * 4)
#define SMEM_CNT_OFF  (SMEM_THR_OFF + MROWS * 4)
#define SMEM_UIN_OFF  (SMEM_CNT_OFF + MROWS * 4)
#define SMEM_FLAG_OFF (SMEM_UIN_OFF + MROWS * 4)
#define SMEM_TOTAL    (SMEM_FLAG_OFF + 16)

// ---------------- kernel 2 (primary): bf16-cache WMMA sims + streaming top-k ----------------
// A tile staged into LDS in WMMA-fragment order via async-to-LDS copies.
__global__ void __launch_bounds__(256) simtopk_bf16_kernel(
    const unsigned short* __restrict__ userb, const unsigned short* __restrict__ cacheb,
    const float* __restrict__ uinv, const float* __restrict__ cinv,
    float* __restrict__ pval, int* __restrict__ pidx) {
  extern __shared__ char smem[];
  float* bufv = (float*)(smem + SMEM_BUFV_OFF);
  int*   bufi = (int*)(smem + SMEM_BUFI_OFF);
  float* thr  = (float*)(smem + SMEM_THR_OFF);
  int*   cnt  = (int*)(smem + SMEM_CNT_OFF);
  float* uin  = (float*)(smem + SMEM_UIN_OFF);
  int*   flg  = (int*)(smem + SMEM_FLAG_OFF);

  const int tid  = threadIdx.x;
  const int lane = tid & 31;
  const int wave = tid >> 5;
  const int wm   = wave & 3;    // 16-row mtile in block
  const int wn   = wave >> 2;   // 16-col ntile in iteration
  const int r15  = lane & 15;
  const int hi   = lane >> 4;
  const int m0   = blockIdx.x * MROWS;
  const int c0   = blockIdx.y * CHUNK;
  const int koB  = hi ? 8 : 0;

  // Async-stage the 64x512 bf16 A tile into LDS, permuted into WMMA fragment
  // order: per (row, kb-block): [K 0..7 | K 16..23] then [K 8..15 | K 24..31].
  // Each 16B source chunk maps to one 16B LDS slot, so per-lane async B128
  // copies implement the permutation directly. (Dynamic-LDS-only kernel =>
  // our byte offsets are the wave's LDS offsets.)
  for (int e = tid; e < MROWS * 64; e += 256) {
    int r = e >> 6, c = e & 63;
    int kbi = c >> 2, j = c & 3;
    unsigned dst = (unsigned)(r * AROWB + kbi * 64 + (j & 1) * 32 + (j >> 1) * 16);
    unsigned long long src =
        (unsigned long long)(const void*)(userb + (size_t)(m0 + r) * D_DIM + c * 8);
    asm volatile("global_load_async_to_lds_b128 %0, %1, off"
                 :: "v"(dst), "v"(src) : "memory");
  }
  for (int r = tid; r < MROWS; r += 256) {
    uin[r] = uinv[m0 + r]; thr[r] = NEG_INF; cnt[r] = 0;
  }
  if (tid == 0) *flg = 0;
  asm volatile("s_wait_asynccnt 0" ::: "memory");
  __syncthreads();

  float uir[8], thrreg[8];
#pragma unroll
  for (int v = 0; v < 8; ++v) { uir[v] = uin[wm * 16 + v + hi * 8]; thrreg[v] = NEG_INF; }

  const char* afrag = smem + (size_t)(wm * 16 + r15) * AROWB + hi * 32;

  for (int it = 0; it < CHUNK / 32; ++it) {        // 125 iterations of 32 cols
    const int ncol = c0 + it * 32 + wn * 16 + r15;
    const unsigned short* crow = cacheb + (size_t)ncol * D_DIM + koB;
    __builtin_prefetch(crow + 32 * D_DIM, 0, 0);   // next tile -> global_prefetch_b8
    v8f acc = {0.f, 0.f, 0.f, 0.f, 0.f, 0.f, 0.f, 0.f};
    for (int kbi = 0; kbi < 16; ++kbi) {
      v8us a0 = *(const v8us*)(afrag + kbi * 64);
      v8us a1 = *(const v8us*)(afrag + kbi * 64 + 16);
      v8us b0 = *(const v8us*)(crow + kbi * 32);
      v8us b1 = *(const v8us*)(crow + kbi * 32 + 16);
      v16us au, bu;
#pragma unroll
      for (int i = 0; i < 8; ++i) {
        au[i] = a0[i]; au[i + 8] = a1[i];
        bu[i] = b0[i]; bu[i + 8] = b1[i];
      }
      acc = __builtin_amdgcn_wmma_f32_16x16x32_bf16(
          false, __builtin_bit_cast(v16bf, au),
          false, __builtin_bit_cast(v16bf, bu),
          (short)0, acc, false, false);
    }
    const float ci = cinv[ncol];
#pragma unroll
    for (int v = 0; v < 8; ++v) {
      float s = acc[v] * uir[v] * ci;               // cosine similarity
      int m = wm * 16 + v + hi * 8;
      if (s > thrreg[v]) {                          // register threshold (conservative)
        int p = atomicAdd(&cnt[m], 1);              // ds_add_rtn_u32
        if (p < CAPC) { bufv[m * CAPC + p] = s; bufi[m * CAPC + p] = ncol; }
      }
    }
    if (it & 1) {                                   // capacity check every 2 iters
      __syncthreads();
      if (tid < MROWS && cnt[tid] > CAPC - 64) *flg = 1;
      __syncthreads();
      bool doC = (*flg != 0);
      __syncthreads();
      if (tid == 0) *flg = 0;
      if (doC) compact_rows(bufv, bufi, thr, cnt, wave, lane);
      __syncthreads();
      if (doC) {
#pragma unroll
        for (int v = 0; v < 8; ++v) thrreg[v] = thr[wm * 16 + v + hi * 8];
      }
    }
  }
  __syncthreads();
  compact_rows(bufv, bufi, thr, cnt, wave, lane);   // final exact top-20 per row
  __syncthreads();

  const size_t base = ((size_t)blockIdx.x * NCHUNKS + blockIdx.y) * (MROWS * K_NEIGH);
  for (int e = tid; e < MROWS * K_NEIGH; e += 256) {
    int r = e / K_NEIGH, j = e - r * K_NEIGH;
    pval[base + e] = bufv[r * CAPC + j];
    pidx[base + e] = bufi[r * CAPC + j];
  }
}

// ---------------- kernel 2 (fallback, small ws): on-the-fly bf16 convert ----------------
__global__ void __launch_bounds__(256) simtopk_kernel(
    const float* __restrict__ user, const float* __restrict__ cache,
    const float* __restrict__ uinv, const float* __restrict__ cinv,
    float* __restrict__ pval, int* __restrict__ pidx) {
  extern __shared__ char smem[];
  unsigned short* ldsA = (unsigned short*)smem;
  float* bufv = (float*)(smem + SMEM_BUFV_OFF);
  int*   bufi = (int*)(smem + SMEM_BUFI_OFF);
  float* thr  = (float*)(smem + SMEM_THR_OFF);
  int*   cnt  = (int*)(smem + SMEM_CNT_OFF);
  float* uin  = (float*)(smem + SMEM_UIN_OFF);
  int*   flg  = (int*)(smem + SMEM_FLAG_OFF);

  const int tid  = threadIdx.x;
  const int lane = tid & 31;
  const int wave = tid >> 5;
  const int wm   = wave & 3;
  const int wn   = wave >> 2;
  const int r15  = lane & 15;
  const int hi   = lane >> 4;
  const int m0   = blockIdx.x * MROWS;
  const int c0   = blockIdx.y * CHUNK;
  const int koA  = hi ? 16 : 0;
  const int koB  = hi ? 16 : 0;

  for (int e = tid; e < MROWS * D_DIM; e += 256) {
    int r = e >> 9, c = e & (D_DIM - 1);
    ldsA[r * ASTRIDE + c] = f2bf(user[(size_t)(m0 + r) * D_DIM + c]);
  }
  for (int r = tid; r < MROWS; r += 256) {
    uin[r] = uinv[m0 + r]; thr[r] = NEG_INF; cnt[r] = 0;
  }
  if (tid == 0) *flg = 0;
  __syncthreads();

  float uir[8], thrreg[8];
#pragma unroll
  for (int v = 0; v < 8; ++v) { uir[v] = uin[wm * 16 + v + hi * 8]; thrreg[v] = NEG_INF; }

  const int arow = (wm * 16 + r15) * ASTRIDE;

  for (int it = 0; it < CHUNK / 32; ++it) {
    const int ncol = c0 + it * 32 + wn * 16 + r15;
    const float* crow = cache + (size_t)ncol * D_DIM + koB;
    __builtin_prefetch(crow + 32 * D_DIM, 0, 0);
    v8f acc = {0.f, 0.f, 0.f, 0.f, 0.f, 0.f, 0.f, 0.f};
    for (int kb = 0; kb < D_DIM; kb += 32) {
      v8us a0 = *(const v8us*)&ldsA[arow + kb + koA];
      v8us a1 = *(const v8us*)&ldsA[arow + kb + koA + 8];
      v16us au;
#pragma unroll
      for (int i = 0; i < 8; ++i) { au[i] = a0[i]; au[i + 8] = a1[i]; }
      const float4* bp = (const float4*)(crow + kb);
      float4 f0 = bp[0], f1 = bp[1], f2 = bp[2], f3 = bp[3];
      v16us bu;
      bu[0]  = f2bf(f0.x); bu[1]  = f2bf(f0.y); bu[2]  = f2bf(f0.z); bu[3]  = f2bf(f0.w);
      bu[4]  = f2bf(f1.x); bu[5]  = f2bf(f1.y); bu[6]  = f2bf(f1.z); bu[7]  = f2bf(f1.w);
      bu[8]  = f2bf(f2.x); bu[9]  = f2bf(f2.y); bu[10] = f2bf(f2.z); bu[11] = f2bf(f2.w);
      bu[12] = f2bf(f3.x); bu[13] = f2bf(f3.y); bu[14] = f2bf(f3.z); bu[15] = f2bf(f3.w);
      acc = __builtin_amdgcn_wmma_f32_16x16x32_bf16(
          false, __builtin_bit_cast(v16bf, au),
          false, __builtin_bit_cast(v16bf, bu),
          (short)0, acc, false, false);
    }
    const float ci = cinv[ncol];
#pragma unroll
    for (int v = 0; v < 8; ++v) {
      float s = acc[v] * uir[v] * ci;
      int m = wm * 16 + v + hi * 8;
      if (s > thrreg[v]) {
        int p = atomicAdd(&cnt[m], 1);
        if (p < CAPC) { bufv[m * CAPC + p] = s; bufi[m * CAPC + p] = ncol; }
      }
    }
    if (it & 1) {
      __syncthreads();
      if (tid < MROWS && cnt[tid] > CAPC - 64) *flg = 1;
      __syncthreads();
      bool doC = (*flg != 0);
      __syncthreads();
      if (tid == 0) *flg = 0;
      if (doC) compact_rows(bufv, bufi, thr, cnt, wave, lane);
      __syncthreads();
      if (doC) {
#pragma unroll
        for (int v = 0; v < 8; ++v) thrreg[v] = thr[wm * 16 + v + hi * 8];
      }
    }
  }
  __syncthreads();
  compact_rows(bufv, bufi, thr, cnt, wave, lane);
  __syncthreads();

  const size_t base = ((size_t)blockIdx.x * NCHUNKS + blockIdx.y) * (MROWS * K_NEIGH);
  for (int e = tid; e < MROWS * K_NEIGH; e += 256) {
    int r = e / K_NEIGH, j = e - r * K_NEIGH;
    pval[base + e] = bufv[r * CAPC + j];
    pidx[base + e] = bufi[r * CAPC + j];
  }
}

// ---------------- kernel 3: merge partials, softmax, aggregate, build combined ----------------
__global__ void __launch_bounds__(256) merge_agg_kernel(
    const float* __restrict__ pval, const int* __restrict__ pidx,
    const float* __restrict__ cache, const float* __restrict__ user,
    float* __restrict__ comb, int* __restrict__ anyv) {
  const int lane = threadIdx.x & 31;
  const int b = blockIdx.x * 8 + (threadIdx.x >> 5);
  if (b >= B_USERS) return;
  const int mg = b >> 6, r = b & (MROWS - 1);
  const int TOT = NCHUNKS * K_NEIGH;                // 500 candidates
  float lv[16]; int li[16];
#pragma unroll
  for (int t = 0; t < 16; ++t) {
    int s = lane + 32 * t;
    if (s < TOT) {
      int c = s / K_NEIGH, j = s - c * K_NEIGH;
      size_t o = ((size_t)mg * NCHUNKS + c) * (MROWS * K_NEIGH) + (size_t)r * K_NEIGH + j;
      lv[t] = pval[o]; li[t] = pidx[o];
    } else { lv[t] = NEG_INF; li[t] = -1; }
  }
  float v0 = NEG_INF, sumExp = 0.f, myv = NEG_INF; int myi = -1;
  for (int k = 0; k < K_NEIGH; ++k) {
    float bv = NEG_INF; int bt = -1;
#pragma unroll
    for (int t = 0; t < 16; ++t) { if (lv[t] > bv) { bv = lv[t]; bt = t; } }
    float rv = bv; int rl = lane;
#pragma unroll
    for (int off = 16; off > 0; off >>= 1) {
      float ov = __shfl_xor(rv, off, 32);
      int   ol = __shfl_xor(rl, off, 32);
      if (ov > rv || (ov == rv && ol < rl)) { rv = ov; rl = ol; }
    }
    int mi = -1;
#pragma unroll
    for (int t = 0; t < 16; ++t) { if (t == bt) mi = li[t]; }
    int gi = __shfl(mi, rl, 32);
    if (lane == rl) {
#pragma unroll
      for (int t = 0; t < 16; ++t) { if (t == bt) lv[t] = NEG_INF; }
    }
    if (k == 0) v0 = rv;
    if (rv >= 0.f) sumExp += expf(rv - v0);         // SIM_THRESHOLD = 0.0
    if (lane == k) { myv = rv; myi = gi; }
  }
  const bool any = (v0 >= 0.f);
  float myw = (lane < K_NEIGH && myv >= 0.f && any) ? (expf(myv - v0) / sumExp) : 0.f;
  float acc[16];
#pragma unroll
  for (int t = 0; t < 16; ++t) acc[t] = 0.f;
  for (int k = 0; k < K_NEIGH; ++k) {
    float wk = __shfl(myw, k, 32);
    int   ik = __shfl(myi, k, 32);
    if (wk > 0.f && ik >= 0) {
      const float* cr = cache + (size_t)ik * D_DIM;
#pragma unroll
      for (int t = 0; t < 16; ++t) acc[t] += wk * cr[lane + 32 * t];
    }
  }
  float* cb = comb + (size_t)b * (2 * D_DIM);
  const float* ur = user + (size_t)b * D_DIM;
#pragma unroll
  for (int t = 0; t < 16; ++t) {
    int d = lane + 32 * t;
    cb[d] = ur[d];
    cb[D_DIM + d] = acc[t];
  }
  if (lane == 0) anyv[b] = any ? 1 : 0;
}

// ---------------- kernels 4/5: fusion MLP with fp32 WMMA 16x16x4 ----------------
__global__ void __launch_bounds__(128) gemm_relu_kernel(
    const float* __restrict__ A, const float* __restrict__ W,
    const float* __restrict__ bias, float* __restrict__ H) {
  const int lane = threadIdx.x & 31, wave = threadIdx.x >> 5;
  const int r15 = lane & 15, hi = lane >> 4;
  const int m0 = blockIdx.x * 16;
  const int ncol = blockIdx.y * 64 + wave * 16 + r15;
  const int KDIM = 2 * D_DIM;
  const float* arow = A + (size_t)(m0 + r15) * KDIM;
  const float* bcol = W + ncol;
  const int ko = hi ? 2 : 0;
  v8f acc = {0.f, 0.f, 0.f, 0.f, 0.f, 0.f, 0.f, 0.f};
#if __has_builtin(__builtin_amdgcn_wmma_f32_16x16x4_f32)
  for (int k = 0; k < KDIM; k += 4) {
    v2f av = *(const v2f*)(arow + k + ko);
    v2f bv; bv[0] = bcol[(size_t)(k + ko) * D_DIM]; bv[1] = bcol[(size_t)(k + ko + 1) * D_DIM];
    acc = __builtin_amdgcn_wmma_f32_16x16x4_f32(false, av, false, bv, (short)0, acc, false, false);
  }
#else
#pragma unroll
  for (int v = 0; v < 8; ++v) {
    const float* am = A + (size_t)(m0 + v + hi * 8) * KDIM;
    float s = 0.f;
    for (int k = 0; k < KDIM; ++k) s += am[k] * W[(size_t)k * D_DIM + ncol];
    acc[v] = s;
  }
#endif
  const float bz = bias[ncol];
#pragma unroll
  for (int v = 0; v < 8; ++v) {
    int mr = m0 + v + hi * 8;
    H[(size_t)mr * D_DIM + ncol] = fmaxf(acc[v] + bz, 0.f);
  }
}

__global__ void __launch_bounds__(128) gemm_out_kernel(
    const float* __restrict__ H, const float* __restrict__ W,
    const float* __restrict__ bias, const int* __restrict__ anyv,
    const float* __restrict__ user, float* __restrict__ out) {
  const int lane = threadIdx.x & 31, wave = threadIdx.x >> 5;
  const int r15 = lane & 15, hi = lane >> 4;
  const int m0 = blockIdx.x * 16;
  const int ncol = blockIdx.y * 64 + wave * 16 + r15;
  const float* arow = H + (size_t)(m0 + r15) * D_DIM;
  const float* bcol = W + ncol;
  const int ko = hi ? 2 : 0;
  v8f acc = {0.f, 0.f, 0.f, 0.f, 0.f, 0.f, 0.f, 0.f};
#if __has_builtin(__builtin_amdgcn_wmma_f32_16x16x4_f32)
  for (int k = 0; k < D_DIM; k += 4) {
    v2f av = *(const v2f*)(arow + k + ko);
    v2f bv; bv[0] = bcol[(size_t)(k + ko) * D_DIM]; bv[1] = bcol[(size_t)(k + ko + 1) * D_DIM];
    acc = __builtin_amdgcn_wmma_f32_16x16x4_f32(false, av, false, bv, (short)0, acc, false, false);
  }
#else
#pragma unroll
  for (int v = 0; v < 8; ++v) {
    const float* am = H + (size_t)(m0 + v + hi * 8) * D_DIM;
    float s = 0.f;
    for (int k = 0; k < D_DIM; ++k) s += am[k] * W[(size_t)k * D_DIM + ncol];
    acc[v] = s;
  }
#endif
  const float bz = bias[ncol];
#pragma unroll
  for (int v = 0; v < 8; ++v) {
    int mr = m0 + v + hi * 8;
    float x = acc[v] + bz;
    out[(size_t)mr * D_DIM + ncol] = anyv[mr] ? x : user[(size_t)mr * D_DIM + ncol];
  }
}

// ---------------- host launch ----------------
extern "C" void kernel_launch(void* const* d_in, const int* in_sizes, int n_in,
                              void* d_out, int out_size, void* d_ws, size_t ws_size,
                              hipStream_t stream) {
  (void)in_sizes; (void)n_in; (void)out_size;
  const float* user  = (const float*)d_in[0];
  const float* cache = (const float*)d_in[1];
  const float* W1    = (const float*)d_in[2];
  const float* b1    = (const float*)d_in[3];
  const float* W2    = (const float*)d_in[4];
  const float* b2    = (const float*)d_in[5];
  float* out = (float*)d_out;
  char*  ws  = (char*)d_ws;

  size_t off = 0;
  auto take = [&](size_t bytes) { size_t r = off; off += (bytes + 255) & ~(size_t)255; return r; };
  float* cinv = (float*)(ws + take((size_t)N_CACHE * 4));
  float* uinv = (float*)(ws + take((size_t)B_USERS * 4));
  float* pval = (float*)(ws + take((size_t)MGROUPS * NCHUNKS * MROWS * K_NEIGH * 4));
  int*   pidx = (int*)  (ws + take((size_t)MGROUPS * NCHUNKS * MROWS * K_NEIGH * 4));
  float* comb = (float*)(ws + take((size_t)B_USERS * 2 * D_DIM * 4));
  float* hbuf = (float*)(ws + take((size_t)B_USERS * D_DIM * 4));
  int*   anyv = (int*)  (ws + take((size_t)B_USERS * 4));
  unsigned short* cacheb = (unsigned short*)(ws + take((size_t)N_CACHE * D_DIM * 2));
  unsigned short* userb  = (unsigned short*)(ws + take((size_t)B_USERS * D_DIM * 2));
  const bool big_ws = (off <= ws_size);   // bf16 staging buffers fit?

  invnorm_kernel<<<(N_CACHE + 7) / 8, 256, 0, stream>>>(cache, cinv, N_CACHE);
  invnorm_kernel<<<B_USERS / 8, 256, 0, stream>>>(user, uinv, B_USERS);

  if (big_ws) {
    const long nc = (long)N_CACHE * D_DIM, nu = (long)B_USERS * D_DIM;
    cvt_bf16_kernel<<<(unsigned)((nc / 4 + 255) / 256), 256, 0, stream>>>(cache, cacheb, nc);
    cvt_bf16_kernel<<<(unsigned)((nu / 4 + 255) / 256), 256, 0, stream>>>(user, userb, nu);
    simtopk_bf16_kernel<<<dim3(MGROUPS, NCHUNKS), 256, SMEM_TOTAL, stream>>>(
        userb, cacheb, uinv, cinv, pval, pidx);
  } else {
    simtopk_kernel<<<dim3(MGROUPS, NCHUNKS), 256, SMEM_TOTAL, stream>>>(
        user, cache, uinv, cinv, pval, pidx);
  }

  merge_agg_kernel<<<B_USERS / 8, 256, 0, stream>>>(pval, pidx, cache, user, comb, anyv);

  gemm_relu_kernel<<<dim3(B_USERS / 16, D_DIM / 64), 128, 0, stream>>>(comb, W1, b1, hbuf);
  gemm_out_kernel<<<dim3(B_USERS / 16, D_DIM / 64), 128, 0, stream>>>(hbuf, W2, b2, anyv, user, out);
}
